// LayerNormLSTM_14181982012189
// MI455X (gfx1250) — matrix operand order repaired and checked by reading, not compiled, not run on hardware
//
#include <hip/hip_runtime.h>

// ---------------------------------------------------------------------------
// LayerNorm-LSTM for MI455X (gfx1250, wave32, WMMA).  S=2048, B=1, I=H=2048.
//
//  - bf16 weights: w_hh = 32MB -> resident in the 192MB L2 across all steps.
//  - Input projection: v_wmma_f32_16x16x32_bf16 GEMM (68.7 GFLOP), parallel.
//  - Recurrence: ONE persistent kernel (64 blocks x 256 = 512 waves), 2048
//    steps, only TWO grid barriers per step:
//      GEMV(+gate-LN tile partials) -> barrier -> per-block LN stats via
//      wave shfl reductions, REDUNDANT elementwise phase (each block computes
//      all 2048 elems, so cz stats need no cross-block exchange), slice
//      writeback of h/c/y -> barrier.
//    h staged in LDS (ds_load_b128 B-fragments); next xg row prefetched
//    (global_prefetch_b8). Fixed-order reductions -> bitwise deterministic.
//  - 6 graph nodes total; barrier count 4096 (was 6144, was ~4100 launches).
// ---------------------------------------------------------------------------

typedef __bf16 bf16_t;
typedef __attribute__((ext_vector_type(16))) __bf16 v16bf;
typedef __attribute__((ext_vector_type(8)))  float  v8f;

#define S_DIM 2048
#define I_DIM 2048
#define H_DIM 2048
#define G_DIM 8192          // 4*H
#define LN_EPS 1e-5f
#define NBLK  64            // persistent grid: 64 blocks x 256 thr = 512 waves
#define TPB   256

union FragBF {
    uint4 u4[2];            // two 16-byte loads
    v16bf v;                // 16 bf16 across 8 VGPRs
};

// -------------------------------- converts --------------------------------

__global__ void k_f32_to_bf16(const float* __restrict__ in,
                              bf16_t* __restrict__ out, int n) {
    for (int i = blockIdx.x * blockDim.x + threadIdx.x; i < n;
         i += gridDim.x * blockDim.x)
        out[i] = (bf16_t)in[i];
}

__global__ void k_init_state(bf16_t* __restrict__ h, float* __restrict__ c,
                             unsigned* __restrict__ bar) {
    int i = blockIdx.x * blockDim.x + threadIdx.x;
    if (i < H_DIM) { h[i] = (bf16_t)0.0f; c[i] = 0.0f; }
    if (i == 0) *bar = 0u;
}

// ------------------------- input projection (GEMM) -------------------------
// xg[s,g] = sum_i x[s,i]*w_ih[g,i] + b_ih[g] + b_hh[g]
// One wave per 16x16 tile; fragments per documented CDNA5 16-bit layouts.

__global__ __launch_bounds__(256)
void k_gemm_xg(const bf16_t* __restrict__ xb,     // (S, I) bf16
               const bf16_t* __restrict__ wih,    // (4H, I) bf16
               const float*  __restrict__ b_ih,
               const float*  __restrict__ b_hh,
               float* __restrict__ xg) {          // (S, 4H) f32
    const int lane = threadIdx.x & 31;
    const int wid  = (blockIdx.x * blockDim.x + threadIdx.x) >> 5;
    const int TN   = G_DIM / 16;
    const int tm   = wid / TN;
    const int tn   = wid - tm * TN;
    const int half = lane >> 4;
    const int l16  = lane & 15;

    const bf16_t* arow = xb  + (size_t)(tm * 16 + l16) * I_DIM + 8 * half;
    const bf16_t* brow = wih + (size_t)(tn * 16 + l16) * I_DIM + 8 * half;

    v8f acc = {};
    #pragma unroll 4
    for (int k0 = 0; k0 < I_DIM; k0 += 32) {
        FragBF a, b;
        a.u4[0] = *reinterpret_cast<const uint4*>(arow + k0);
        a.u4[1] = *reinterpret_cast<const uint4*>(arow + k0 + 16);
        b.u4[0] = *reinterpret_cast<const uint4*>(brow + k0);
        b.u4[1] = *reinterpret_cast<const uint4*>(brow + k0 + 16);
        acc = __builtin_amdgcn_wmma_f32_16x16x32_bf16(
            false, a.v, false, b.v, (short)0, acc, false, false);
    }
    const int   g    = tn * 16 + l16;
    const float bias = b_ih[g] + b_hh[g];
    float* out = xg + (size_t)(tm * 16 + 8 * half) * G_DIM + g;
    #pragma unroll
    for (int r = 0; r < 8; ++r)
        out[(size_t)r * G_DIM] = acc[r] + bias;
}

// --------------------- persistent recurrence kernel ------------------------

__device__ __forceinline__ void grid_barrier(unsigned* bar, unsigned expected) {
    __syncthreads();
    if (threadIdx.x == 0) {
        __threadfence();                                   // flush my stores
        __hip_atomic_fetch_add(bar, 1u, __ATOMIC_RELEASE,
                               __HIP_MEMORY_SCOPE_AGENT);
        while (__hip_atomic_load(bar, __ATOMIC_ACQUIRE,
                                 __HIP_MEMORY_SCOPE_AGENT) < expected)
            __builtin_amdgcn_s_sleep(1);
    }
    __syncthreads();
    __threadfence();                                       // invalidate stale
}

__global__ __launch_bounds__(TPB)
void k_lstm_persist(const bf16_t* __restrict__ whh,   // (4H, H) bf16
                    const float*  __restrict__ xg,    // (S, 4H) f32
                    const float*  __restrict__ gamma, // (5, H)
                    const float*  __restrict__ beta,  // (5, H)
                    float*  __restrict__ gates,       // (4H)
                    float*  __restrict__ gpart,       // (512) tile sums
                    float*  __restrict__ gpart2,      // (512) tile sumsq
                    float*  __restrict__ c,           // (H) f32 state
                    bf16_t* __restrict__ h,           // (H) bf16 state
                    float*  __restrict__ y,           // (S, H)
                    float*  __restrict__ hy_out,      // (H)
                    float*  __restrict__ cy_out,      // (H)
                    unsigned* __restrict__ bar) {
    __shared__ float  red[16];          // 8 wave sums + 8 wave sumsq (cz)
    __shared__ float  stats[8];         // {mean,rstd} x 4 gates
    __shared__ bf16_t hs[H_DIM];        // staged h (4KB)

    const int tid  = threadIdx.x;
    const int lane = tid & 31;
    const int wv   = tid >> 5;
    const int blk  = blockIdx.x;
    const int tile = blk * 8 + wv;      // 0..511 -> 16 gate rows
    const int half = lane >> 4;
    const int l16  = lane & 15;
    const int row0 = tile * 16 + 8 * half;
    // output slice: wave (blk&7) of this block writes e = blk*32 + t
    const int ts0  = (blk & 7) * 32;    // first thread of the slice-owner wave
    const int jsl  = blk >> 3;          // j index of the slice elements

    const bf16_t* arow = whh + (size_t)(tile * 16 + l16) * H_DIM + 8 * half;

    unsigned epoch = 0;
    for (int s = 0; s < S_DIM; ++s) {
        // ---- stage h into LDS (512 dwords, 2 per thread) ----
        {
            const unsigned* hg = (const unsigned*)h;
            unsigned* hl = (unsigned*)hs;
            hl[tid]       = hg[tid];
            hl[tid + 256] = hg[tid + 256];
        }
        if (lane == 0 && s + 1 < S_DIM)       // warm L2/L0 for next xg row
            __builtin_prefetch(xg + (size_t)(s + 1) * G_DIM + tile * 16, 0, 3);
        __syncthreads();

        // ---- WMMA GEMV: 16 gate rows per wave, B = h broadcast ----
        v8f acc = {};
        #pragma unroll 4
        for (int k0 = 0; k0 < H_DIM; k0 += 32) {
            FragBF a, b;
            a.u4[0] = *reinterpret_cast<const uint4*>(arow + k0);
            a.u4[1] = *reinterpret_cast<const uint4*>(arow + k0 + 16);
            const bf16_t* hbk = hs + 8 * half + k0;
            b.u4[0] = *reinterpret_cast<const uint4*>(hbk);      // ds_load
            b.u4[1] = *reinterpret_cast<const uint4*>(hbk + 16);
            acc = __builtin_amdgcn_wmma_f32_16x16x32_bf16(
                false, a.v, false, b.v, (short)0, acc, false, false);
        }
        // Every lane of a half holds the same 8 row values (B cols identical).
        const float* xgrow = xg + (size_t)s * G_DIM;
        float vals[8], p1 = 0.f, p2 = 0.f;
        #pragma unroll
        for (int r = 0; r < 8; ++r) {
            const float v = acc[r] + xgrow[row0 + r];
            vals[r] = v; p1 += v; p2 += v * v;
        }
        p1 += __shfl_xor(p1, 16);       // combine the two halves
        p2 += __shfl_xor(p2, 16);
        if (lane == 0) { gpart[tile] = p1; gpart2[tile] = p2; }
        if (l16 == 0) {
            #pragma unroll
            for (int r = 0; r < 8; ++r) gates[row0 + r] = vals[r];
        }
        grid_barrier(bar, ++epoch * NBLK);                 // gates ready

        // ---- gate LN stats: waves 0..3 reduce 128 tile-partials each ----
        if (wv < 4) {
            float s1 = 0.f, s2 = 0.f;
            #pragma unroll
            for (int k = 0; k < 4; ++k) {
                s1 += gpart [wv * 128 + lane + 32 * k];
                s2 += gpart2[wv * 128 + lane + 32 * k];
            }
            #pragma unroll
            for (int o = 16; o > 0; o >>= 1) {
                s1 += __shfl_xor(s1, o);
                s2 += __shfl_xor(s2, o);
            }
            if (lane == 0) {
                const float mean = s1 * (1.0f / H_DIM);
                const float var  = s2 * (1.0f / H_DIM) - mean * mean;
                stats[2 * wv]     = mean;
                stats[2 * wv + 1] = rsqrtf(var + LN_EPS);
            }
        }
        __syncthreads();

        // ---- elementwise phase: REDUNDANT per block (8 elems/thread) ----
        float czv[8], ov[8], q1 = 0.f, q2 = 0.f;
        #pragma unroll
        for (int j = 0; j < 8; ++j) {
            const int e = tid + 256 * j;
            float a4[4];
            #pragma unroll
            for (int q = 0; q < 4; ++q) {
                const float z  = gates[q * H_DIM + e];
                const float ln = (z - stats[2 * q]) * stats[2 * q + 1]
                                 * gamma[q * H_DIM + e] + beta[q * H_DIM + e];
                a4[q] = (q == 2) ? tanhf(ln) : (1.0f / (1.0f + __expf(-ln)));
            }
            const float cz = a4[1] * c[e] + a4[0] * a4[2];
            czv[j] = cz; ov[j] = a4[3];
            q1 += cz; q2 += cz * cz;
        }
        // block-reduce cz stats: wave shfl + 16-slot LDS (fixed order)
        #pragma unroll
        for (int o = 16; o > 0; o >>= 1) {
            q1 += __shfl_xor(q1, o);
            q2 += __shfl_xor(q2, o);
        }
        if (lane == 0) { red[wv] = q1; red[8 + wv] = q2; }
        __syncthreads();
        float csum = 0.f, csum2 = 0.f;
        #pragma unroll
        for (int w = 0; w < 8; ++w) { csum += red[w]; csum2 += red[8 + w]; }
        const float cmean = csum * (1.0f / H_DIM);
        const float cvar  = csum2 * (1.0f / H_DIM) - cmean * cmean;
        const float crstd = rsqrtf(cvar + LN_EPS);

        // ---- slice writeback: wave (blk&7) writes e = blk*32 + t ----
        if (tid >= ts0 && tid < ts0 + 32) {
            const int e  = tid + 256 * jsl;     // == blk*32 + (tid - ts0)
            const float cy = (czv[jsl] - cmean) * crstd
                             * gamma[4 * H_DIM + e] + beta[4 * H_DIM + e];
            const float hyv = ov[jsl] * tanhf(cy);
            c[e] = cy;
            h[e] = (bf16_t)hyv;
            y[(size_t)s * H_DIM + e] = hyv;
            if (s == S_DIM - 1) { hy_out[e] = hyv; cy_out[e] = cy; }
        }
        grid_barrier(bar, ++epoch * NBLK);                 // h/c ready
    }
}

// --------------------------------- launch ----------------------------------

extern "C" void kernel_launch(void* const* d_in, const int* in_sizes, int n_in,
                              void* d_out, int out_size, void* d_ws, size_t ws_size,
                              hipStream_t stream) {
    const float* x        = (const float*)d_in[0];   // (S, 1, I)
    const float* w_ih     = (const float*)d_in[1];   // (4H, I)
    const float* w_hh     = (const float*)d_in[2];   // (4H, H)
    const float* b_ih     = (const float*)d_in[3];   // (4H)
    const float* b_hh     = (const float*)d_in[4];   // (4H)
    const float* ln_gamma = (const float*)d_in[5];   // (5, H)
    const float* ln_beta  = (const float*)d_in[6];   // (5, H)
    float* out = (float*)d_out;                      // y (S*H) | hy (H) | cy (H)

    char* ws = (char*)d_ws;
    size_t off = 0;
    auto wsalloc = [&](size_t bytes) -> void* {
        void* p = ws + off;
        off += (bytes + 255) & ~(size_t)255;
        return p;
    };
    bf16_t*   xb     = (bf16_t*)wsalloc((size_t)S_DIM * I_DIM * sizeof(bf16_t));
    bf16_t*   wihb   = (bf16_t*)wsalloc((size_t)G_DIM * I_DIM * sizeof(bf16_t));
    bf16_t*   whhb   = (bf16_t*)wsalloc((size_t)G_DIM * H_DIM * sizeof(bf16_t));
    float*    xg     = (float*) wsalloc((size_t)S_DIM * G_DIM * sizeof(float));
    float*    gates  = (float*) wsalloc((size_t)G_DIM * sizeof(float));
    float*    gpart  = (float*) wsalloc(512 * sizeof(float));
    float*    gpart2 = (float*) wsalloc(512 * sizeof(float));
    float*    c      = (float*) wsalloc((size_t)H_DIM * sizeof(float));
    bf16_t*   h      = (bf16_t*)wsalloc((size_t)H_DIM * sizeof(bf16_t));
    unsigned* bar    = (unsigned*)wsalloc(sizeof(unsigned));

    // 1) bf16 conversions + zero state/barrier (every call: deterministic).
    k_f32_to_bf16<<<2048, 256, 0, stream>>>(x,    xb,   S_DIM * I_DIM);
    k_f32_to_bf16<<<2048, 256, 0, stream>>>(w_ih, wihb, G_DIM * I_DIM);
    k_f32_to_bf16<<<2048, 256, 0, stream>>>(w_hh, whhb, G_DIM * H_DIM);
    k_init_state<<<(H_DIM + 255) / 256, 256, 0, stream>>>(h, c, bar);

    // 2) Input projection GEMM: 65536 waves, 8 waves/block.
    k_gemm_xg<<<(S_DIM / 16) * (G_DIM / 16) / 8, 256, 0, stream>>>(
        xb, wihb, b_ih, b_hh, xg);

    // 3) Whole recurrence in one persistent kernel (64 co-resident blocks).
    float* hy_out = out + (size_t)S_DIM * H_DIM;
    float* cy_out = hy_out + H_DIM;
    k_lstm_persist<<<NBLK, TPB, 0, stream>>>(
        whhb, xg, ln_gamma, ln_beta, gates, gpart, gpart2,
        c, h, out, hy_out, cy_out, bar);
}
